// LSTMModel_53094385713936
// MI455X (gfx1250) — matrix operand order repaired and checked by reading, not compiled, not run on hardware
//
#include <hip/hip_runtime.h>

#define B_ 128
#define S_ 256
#define E_ 512
#define H_ 1024
#define V_ 2048
#define G4H (4 * H_)

typedef __bf16 bf16;
typedef bf16 bf16x8  __attribute__((ext_vector_type(8)));
typedef bf16 bf16x16 __attribute__((ext_vector_type(16)));
typedef float f32x8  __attribute__((ext_vector_type(8)));

// -------------------- utility kernels --------------------

__global__ void zero_kernel(float4* __restrict__ p, long n16) {
    long i = blockIdx.x * (long)blockDim.x + threadIdx.x;
    float4 z = {0.f, 0.f, 0.f, 0.f};
    for (; i < n16; i += (long)gridDim.x * blockDim.x) p[i] = z;
}

// f32 src [R,C] row-major -> bf16 dst [C,R] row-major (transpose + convert)
__global__ void convert_transpose_kernel(const float* __restrict__ src,
                                         bf16* __restrict__ dst, int R, int C) {
    long total = (long)R * C;
    for (long i = blockIdx.x * (long)blockDim.x + threadIdx.x; i < total;
         i += (long)gridDim.x * blockDim.x) {
        int r = (int)(i / C);
        int c = (int)(i % C);
        dst[(long)c * R + r] = (bf16)src[i];
    }
}

// Embedding gather into xe[S*B, E] bf16, row = s*B + b (the 'sbg' layout)
__global__ void embed_kernel(const int* __restrict__ tokens,
                             const float* __restrict__ emb,
                             bf16* __restrict__ xe) {
    int row = blockIdx.x;            // 0 .. S*B-1
    int s = row / B_;
    int b = row % B_;
    int tok = tokens[b * S_ + s];
    const float* e = emb + (long)tok * E_;
    bf16* o = xe + (long)row * E_;
    for (int j = threadIdx.x; j < E_; j += blockDim.x) o[j] = (bf16)e[j];
}

// -------------------- bf16 WMMA GEMM --------------------
// D[M,N](f32) = A[M,K](bf16,row-major) x Bt[N,K](bf16, pre-transposed)
//               (+ bias[N]) (+ Cadd[M,N])
// blockIdx.z selects between two independent problems (fwd/bwd direction).
// Requires M%128==0, N%128==0, K%32==0.
//
// Staging uses CDNA5 GLOBAL_LOAD_ASYNC_TO_LDS_B128 (ASYNCcnt) with
// double-buffered LDS so the DMA for tile k+1 overlaps the WMMAs on tile k.

#define BLK 128
#define BK  32
#define LDP (BK + 8)   // padded LDS row stride in bf16 elems (80B rows, 16B aligned)

__device__ __forceinline__ unsigned lds_off32(const void* p) {
    // Generic pointers into LDS are {SHARED_BASE[63:32], lds_offset[31:0]}
    // (ISA 10.2 aperture rules), so the low 32 bits are the DS byte address.
    return (unsigned)(unsigned long long)(uintptr_t)p;
}

__device__ __forceinline__ void async_b128_to_lds(const bf16* gptr, const bf16* lptr) {
    unsigned l = lds_off32(lptr);
    unsigned long long g = (unsigned long long)(uintptr_t)gptr;
    asm volatile("global_load_async_to_lds_b128 %0, %1, off"
                 :: "v"(l), "v"(g) : "memory");
}

__device__ __forceinline__ void wait_async0() {
    asm volatile("s_wait_asynccnt 0" ::: "memory");
}

__global__ __launch_bounds__(256) void gemm_bf16_wmma(
    const bf16* __restrict__ A0, const bf16* __restrict__ Bt0,
    const float* __restrict__ Cadd0, const float* __restrict__ bias0,
    float* __restrict__ D0,
    const bf16* __restrict__ A1, const bf16* __restrict__ Bt1,
    const float* __restrict__ Cadd1, const float* __restrict__ bias1,
    float* __restrict__ D1,
    int M, int N, int K)
{
    const bf16*  A  = blockIdx.z ? A1 : A0;
    const bf16*  Bt = blockIdx.z ? Bt1 : Bt0;
    const float* Ca = blockIdx.z ? Cadd1 : Cadd0;
    const float* bi = blockIdx.z ? bias1 : bias0;
    float*       D  = blockIdx.z ? D1 : D0;

    __shared__ __align__(16) bf16 sA[2][BLK * LDP];
    __shared__ __align__(16) bf16 sB[2][BLK * LDP];

    const int tid  = threadIdx.x;
    const int lane = tid & 31;
    const int wave = tid >> 5;
    const int wm   = wave & 1;    // which 64-row half of the block tile
    const int wn   = wave >> 1;   // which 32-col quarter

    const int m0 = blockIdx.y * BLK;
    const int n0 = blockIdx.x * BLK;

    // Per-thread staging chunks: 2x (A,B) 16B chunks -> 512 chunks per tile.
    const int r0  = tid >> 2;
    const int kc0 = (tid & 3) * 8;
    const int r1  = (tid + 256) >> 2;
    const int kc1 = ((tid + 256) & 3) * 8;

    f32x8 acc[4][2];
#pragma unroll
    for (int mi = 0; mi < 4; ++mi)
#pragma unroll
        for (int ni = 0; ni < 2; ++ni)
#pragma unroll
            for (int v = 0; v < 8; ++v) acc[mi][ni][v] = 0.f;

    const int rA  = lane & 15;     // row/col within 16x16 fragment
    const int hiA = lane >> 4;     // lane-half selects K sub-range

    // Prologue: DMA tile 0 into buffer 0.
    {
        const int kb = 0;
        async_b128_to_lds(A  + (long)(m0 + r0) * K + kb + kc0, &sA[0][r0 * LDP + kc0]);
        async_b128_to_lds(Bt + (long)(n0 + r0) * K + kb + kc0, &sB[0][r0 * LDP + kc0]);
        async_b128_to_lds(A  + (long)(m0 + r1) * K + kb + kc1, &sA[0][r1 * LDP + kc1]);
        async_b128_to_lds(Bt + (long)(n0 + r1) * K + kb + kc1, &sB[0][r1 * LDP + kc1]);
        wait_async0();
        __syncthreads();
    }

    const int nk = K / BK;
    for (int ik = 0; ik < nk; ++ik) {
        const int cur = ik & 1;
        const int nxt = cur ^ 1;

        // Kick off DMA for the next tile into the other buffer (overlaps WMMAs).
        if (ik + 1 < nk) {
            const int kb = (ik + 1) * BK;
            async_b128_to_lds(A  + (long)(m0 + r0) * K + kb + kc0, &sA[nxt][r0 * LDP + kc0]);
            async_b128_to_lds(Bt + (long)(n0 + r0) * K + kb + kc0, &sB[nxt][r0 * LDP + kc0]);
            async_b128_to_lds(A  + (long)(m0 + r1) * K + kb + kc1, &sA[nxt][r1 * LDP + kc1]);
            async_b128_to_lds(Bt + (long)(n0 + r1) * K + kb + kc1, &sB[nxt][r1 * LDP + kc1]);
        }

        // A fragments (16x32 bf16, ISA layout): lanes 0-15 hold K[0..7],K[16..23],
        // lanes 16-31 hold K[8..15],K[24..31], two bf16 per VGPR.
        bf16x16 af[4];
#pragma unroll
        for (int mi = 0; mi < 4; ++mi) {
            const bf16* p = &sA[cur][(wm * 64 + mi * 16 + rA) * LDP + hiA * 8];
            bf16x8 lo = *(const bf16x8*)p;
            bf16x8 hi = *(const bf16x8*)(p + 16);
            af[mi] = __builtin_shufflevector(lo, hi, 0, 1, 2, 3, 4, 5, 6, 7,
                                             8, 9, 10, 11, 12, 13, 14, 15);
        }
        // B fragments (32x16): lanes 0-15 hold K[0..7],K[16..23] of column N=lane,
        // lanes 16-31 hold K[8..15],K[24..31].
        bf16x16 bfr[2];
#pragma unroll
        for (int ni = 0; ni < 2; ++ni) {
            const bf16* p = &sB[cur][(wn * 32 + ni * 16 + rA) * LDP + hiA * 8];
            bf16x8 lo = *(const bf16x8*)p;
            bf16x8 hi = *(const bf16x8*)(p + 16);
            bfr[ni] = __builtin_shufflevector(lo, hi, 0, 1, 2, 3, 4, 5, 6, 7,
                                              8, 9, 10, 11, 12, 13, 14, 15);
        }
#pragma unroll
        for (int mi = 0; mi < 4; ++mi)
#pragma unroll
            for (int ni = 0; ni < 2; ++ni)
                acc[mi][ni] = __builtin_amdgcn_wmma_f32_16x16x32_bf16(
                    false, af[mi], false, bfr[ni], (short)0, acc[mi][ni],
                    false, false);

        // All of this wave's DMAs into buf[nxt] landed; barrier publishes them
        // (and guarantees everyone finished reading buf[cur] before reuse).
        wait_async0();
        __syncthreads();
    }

    // Epilogue: C layout = VGPR v -> M = v + 8*(lane>=16), N = lane&15.
    const int colBase = n0 + wn * 32 + (lane & 15);
    const int rowBase = m0 + wm * 64 + (lane >> 4) * 8;
#pragma unroll
    for (int mi = 0; mi < 4; ++mi) {
#pragma unroll
        for (int ni = 0; ni < 2; ++ni) {
            int col = colBase + ni * 16;
            float bv = bi ? bi[col] : 0.f;
#pragma unroll
            for (int v = 0; v < 8; ++v) {
                int row = rowBase + mi * 16 + v;
                float val = acc[mi][ni][v] + bv;
                if (Ca) val += Ca[(long)row * N + col];
                D[(long)row * N + col] = val;
            }
        }
    }
}

// -------------------- LSTM gate pointwise --------------------
// Z[dir] = [B, 4H] pre-activation (xt + h@U + b). Keras gate order i,f,g,o.
// dir 0: forward, tanh activation. dir 1: backward, relu activation.
// Writes next h (bf16 state for next GEMM) and scatters into hcat[B,S,2H].

__device__ __forceinline__ float sigmoidf_(float x) {
    return 1.f / (1.f + __expf(-x));
}

__global__ void lstm_pointwise_kernel(
    const float* __restrict__ Zf, const float* __restrict__ Zb,
    float* __restrict__ cf, float* __restrict__ cb,
    bf16* __restrict__ hf, bf16* __restrict__ hb,
    bf16* __restrict__ hcat, int t)
{
    int dir = blockIdx.y;
    int idx = blockIdx.x * blockDim.x + threadIdx.x;   // 0 .. B*H-1
    int b = idx / H_;
    int j = idx % H_;

    const float* Z = dir ? Zb : Zf;
    float*       c = dir ? cb : cf;
    bf16*        h = dir ? hb : hf;

    long zb = (long)b * G4H;
    float ig = sigmoidf_(Z[zb + j]);
    float fg = sigmoidf_(Z[zb + H_ + j]);
    float gg = Z[zb + 2 * H_ + j];
    float og = sigmoidf_(Z[zb + 3 * H_ + j]);

    float cc, hh;
    if (dir == 0) {
        cc = fg * c[idx] + ig * tanhf(gg);
        hh = og * tanhf(cc);
    } else {
        float g = gg > 0.f ? gg : 0.f;
        cc = fg * c[idx] + ig * g;
        float a = cc > 0.f ? cc : 0.f;
        hh = og * a;
    }
    c[idx] = cc;
    h[idx] = (bf16)hh;

    int s = dir ? (S_ - 1 - t) : t;   // backward scan step t = forward time S-1-t
    hcat[((long)b * S_ + s) * (2 * H_) + (long)dir * H_ + j] = (bf16)hh;
}

// -------------------- launch --------------------

extern "C" void kernel_launch(void* const* d_in, const int* in_sizes, int n_in,
                              void* d_out, int out_size, void* d_ws, size_t ws_size,
                              hipStream_t stream) {
    const int*   tokens = (const int*)d_in[0];
    const float* emb = (const float*)d_in[1];
    const float* Wf  = (const float*)d_in[2];
    const float* Uf  = (const float*)d_in[3];
    const float* bfv = (const float*)d_in[4];
    const float* Wb  = (const float*)d_in[5];
    const float* Ub  = (const float*)d_in[6];
    const float* bbv = (const float*)d_in[7];
    const float* Wd  = (const float*)d_in[8];
    const float* bdv = (const float*)d_in[9];
    float* out = (float*)d_out;

    char* ws = (char*)d_ws;
    size_t off = 0;
    auto alloc = [&](size_t bytes) -> void* {
        void* p = ws + off;
        off += (bytes + 255) & ~(size_t)255;
        return p;
    };

    bf16*  xe   = (bf16*)alloc((size_t)S_ * B_ * E_ * sizeof(bf16));
    bf16*  Wft  = (bf16*)alloc((size_t)G4H * E_ * sizeof(bf16));
    bf16*  Wbt  = (bf16*)alloc((size_t)G4H * E_ * sizeof(bf16));
    bf16*  Uft  = (bf16*)alloc((size_t)G4H * H_ * sizeof(bf16));
    bf16*  Ubt  = (bf16*)alloc((size_t)G4H * H_ * sizeof(bf16));
    bf16*  Wdt  = (bf16*)alloc((size_t)V_ * (2 * H_) * sizeof(bf16));
    float* xwf  = (float*)alloc((size_t)S_ * B_ * G4H * sizeof(float));
    float* xwb  = (float*)alloc((size_t)S_ * B_ * G4H * sizeof(float));
    float* Zf   = (float*)alloc((size_t)B_ * G4H * sizeof(float));
    float* Zb   = (float*)alloc((size_t)B_ * G4H * sizeof(float));
    // state region (zeroed each call): hf, hb (bf16), cf, cb (f32) contiguous
    char*  stateBase = ws + off;
    bf16*  hf = (bf16*)alloc((size_t)B_ * H_ * sizeof(bf16));
    bf16*  hb = (bf16*)alloc((size_t)B_ * H_ * sizeof(bf16));
    float* cf = (float*)alloc((size_t)B_ * H_ * sizeof(float));
    float* cb = (float*)alloc((size_t)B_ * H_ * sizeof(float));
    size_t stateBytes = (size_t)(ws + off - stateBase);
    bf16*  hcat = (bf16*)alloc((size_t)B_ * S_ * (2 * H_) * sizeof(bf16));

    // 1) weight transpose+convert to bf16 [N,K]
    convert_transpose_kernel<<<2048, 256, 0, stream>>>(Wf, Wft, E_, G4H);
    convert_transpose_kernel<<<2048, 256, 0, stream>>>(Wb, Wbt, E_, G4H);
    convert_transpose_kernel<<<2048, 256, 0, stream>>>(Uf, Uft, H_, G4H);
    convert_transpose_kernel<<<2048, 256, 0, stream>>>(Ub, Ubt, H_, G4H);
    convert_transpose_kernel<<<2048, 256, 0, stream>>>(Wd, Wdt, 2 * H_, V_);

    // 2) embedding gather into [S*B, E] bf16
    embed_kernel<<<S_ * B_, 256, 0, stream>>>(tokens, emb, xe);

    // 3) zero h/c state
    zero_kernel<<<256, 256, 0, stream>>>((float4*)stateBase, (long)(stateBytes / 16));

    // 4) input projections (both directions fused): xw = xe @ W + b
    {
        dim3 grid(G4H / BLK, (S_ * B_) / BLK, 2);
        gemm_bf16_wmma<<<grid, 256, 0, stream>>>(
            xe, Wft, nullptr, bfv, xwf,
            xe, Wbt, nullptr, bbv, xwb,
            S_ * B_, G4H, E_);
    }

    // 5) recurrent scan: per step, fused fwd+bwd GEMM then fused gate kernel
    {
        dim3 gGemm(G4H / BLK, B_ / BLK, 2);
        dim3 gPw((B_ * H_) / 256, 2);
        for (int t = 0; t < S_; ++t) {
            const float* caF = xwf + (long)t * B_ * G4H;
            const float* caB = xwb + (long)(S_ - 1 - t) * B_ * G4H;
            gemm_bf16_wmma<<<gGemm, 256, 0, stream>>>(
                hf, Uft, caF, nullptr, Zf,
                hb, Ubt, caB, nullptr, Zb,
                B_, G4H, H_);
            lstm_pointwise_kernel<<<gPw, 256, 0, stream>>>(
                Zf, Zb, cf, cb, hf, hb, hcat, t);
        }
    }

    // 6) output projection: out[B*S, V] = hcat @ Wd + bd
    {
        dim3 grid(V_ / BLK, (B_ * S_) / BLK, 1);
        gemm_bf16_wmma<<<grid, 256, 0, stream>>>(
            hcat, Wdt, nullptr, bdv, out,
            nullptr, nullptr, nullptr, nullptr, nullptr,
            B_ * S_, V_, 2 * H_);
    }
}
